// HubUniformLoss_74268574482706
// MI455X (gfx1250) — compile-verified, matrix operand order checked by reading, not run on hardware
//
#include <hip/hip_runtime.h>

// ---------------------------------------------------------------------------
// Fused Gram-matrix logsumexp for MI455X (gfx1250, wave32, WMMA + TDM).
//   loss = logsumexp_{i,j}( KAPPA * <e_i, e_j> ),  E: 16384 x 128 fp32
// bf16 hi/lo split (3x v_wmma_f32_16x16x32_bf16, ~fp32 accuracy), tiles
// staged into LDS by the Tensor Data Mover (tensor_load_to_lds), streaming
// hierarchical logsumexp (never materialize the 1 GB logits matrix).
// ---------------------------------------------------------------------------

#define KAPPA 20.0f
#define N_PTS 16384
#define D_DIM 128
#define BM 128
#define BN 128
#define LDS_STRIDE 136   // 128 + 8 bf16 pad: row shift = 4 banks -> conflict-light

typedef __bf16 v16bf __attribute__((ext_vector_type(16)));
typedef float  v8f   __attribute__((ext_vector_type(8)));
typedef unsigned int v4u __attribute__((ext_vector_type(4)));
typedef unsigned int v8u __attribute__((ext_vector_type(8)));

// ---------------------------------------------------------------------------
// Kernel 1: split fp32 -> bf16 hi + bf16 lo (x ~= hi + lo)
// ---------------------------------------------------------------------------
__global__ __launch_bounds__(256) void split_bf16_kernel(
    const float* __restrict__ x, unsigned short* __restrict__ hi_u,
    unsigned short* __restrict__ lo_u, int n4) {
  int i = blockIdx.x * 256 + threadIdx.x;
  if (i >= n4) return;
  float4 v = reinterpret_cast<const float4*>(x)[i];
  float vs[4] = {v.x, v.y, v.z, v.w};
  union Pack4 { __bf16 b[4]; uint2 u; } H, L;
#pragma unroll
  for (int k = 0; k < 4; ++k) {
    __bf16 h = (__bf16)vs[k];
    H.b[k] = h;
    L.b[k] = (__bf16)(vs[k] - (float)h);
  }
  reinterpret_cast<uint2*>(hi_u)[i] = H.u;
  reinterpret_cast<uint2*>(lo_u)[i] = L.u;
}

// ---------------------------------------------------------------------------
// TDM: DMA one 128x128 bf16 tile (row stride 128 elems in memory) into LDS
// with 16B padding after each 256B row (-> LDS stride 136 elems).
// D# per CDNA5 ISA ch.8; 2D tile -> groups 2/3 omitted (NULL).
// ---------------------------------------------------------------------------
__device__ __forceinline__ void tdm_load_tile(const __bf16* gsrc,
                                              __bf16* ldst) {
  unsigned long long ga = (unsigned long long)(uintptr_t)gsrc;
  unsigned lds_off = (unsigned)(uintptr_t)ldst;  // low 32 bits = LDS offset

  v4u g0;
  g0[0] = 1u;                                   // count=1, user descriptor
  g0[1] = lds_off;                              // lds_addr
  g0[2] = (unsigned)ga;                         // global_addr[31:0]
  g0[3] = ((unsigned)(ga >> 32) & 0x01FFFFFFu)  // global_addr[56:32]
          | (2u << 30);                         // type=2 ("image")

  v8u g1;
  g1[0] = (1u << 16)    // data_size = 2 bytes
        | (1u << 20)    // pad_enable
        | (5u << 22)    // pad_interval: 64 DWORDs (= one 256B row)
        | (3u << 25);   // pad_amount: 4 DWORDs (= 16B)
  g1[1] = (128u & 0xFFFFu) << 16;        // tensor_dim0[15:0]=128
  g1[2] = 0u;                            // tensor_dim0 hi / tensor_dim1 lo (=0)
  g1[3] = 0x10u | (128u << 16);          // tensor_dim1=0x100000, tile_dim0=128
  g1[4] = 128u;                          // tile_dim1=128, tile_dim2=0
  g1[5] = 128u;                          // tensor_dim0_stride = 128 elems
  g1[6] = 0u;
  g1[7] = 0u;

  asm volatile("tensor_load_to_lds %0, %1" ::"s"(g0), "s"(g1) : "memory");
}

// A-fragment (16x32 bf16, MxK) per CDNA5 ISA layout:
//   lanes 0-15 : M = lane,    K = kbase+[0..7]  (v0-3), kbase+[16..23] (v4-7)
//   lanes 16-31: M = lane-16, K = kbase+[8..15] (v0-3), kbase+[24..31] (v4-7)
// B = E^T, so B fragments (32x16, K-major columns) use identical addressing
// on the column-block's rows of E.
__device__ __forceinline__ v16bf load_frag(const __bf16* lds, int rowBase,
                                           int kbase, int lane) {
  const int r  = rowBase + (lane & 15);
  const int k0 = kbase + ((lane & 16) >> 1);  // +8 for upper half-wave
  union Frag { uint4 q[2]; v16bf v; } f;
  const __bf16* p = lds + r * LDS_STRIDE + k0;
  f.q[0] = *reinterpret_cast<const uint4*>(p);
  f.q[1] = *reinterpret_cast<const uint4*>(p + 16);
  return f.v;
}

// ---------------------------------------------------------------------------
// Kernel 2: one 128x128 logit tile per block, fused partial logsumexp
// ---------------------------------------------------------------------------
__global__ __launch_bounds__(256) void gram_lse_kernel(
    const unsigned short* __restrict__ Ehi_u,
    const unsigned short* __restrict__ Elo_u, float2* __restrict__ partials) {
  __shared__ __bf16 As_hi[BM * LDS_STRIDE];
  __shared__ __bf16 As_lo[BM * LDS_STRIDE];
  __shared__ __bf16 Bs_hi[BN * LDS_STRIDE];
  __shared__ __bf16 Bs_lo[BN * LDS_STRIDE];
  __shared__ float red_m[256];
  __shared__ float red_s[256];

  const __bf16* Ehi = reinterpret_cast<const __bf16*>(Ehi_u);
  const __bf16* Elo = reinterpret_cast<const __bf16*>(Elo_u);

  const int tid  = threadIdx.x;
  const int lane = tid & 31;
  const int wave = tid >> 5;
  const int wm   = wave & 3;  // 4 waves along M -> 32 rows each
  const int wn   = wave >> 2; // 2 waves along N -> 64 cols each

  const size_t rowA0 = (size_t)blockIdx.y * BM;
  const size_t rowB0 = (size_t)blockIdx.x * BN;

  // Wave 0 drives the Tensor Data Mover for all 4 tiles (TDM ignores EXEC,
  // so force a true scalar branch via readfirstlane to avoid duplicate DMAs).
  if (__builtin_amdgcn_readfirstlane(wave) == 0) {
    tdm_load_tile(Ehi + rowA0 * D_DIM, As_hi);
    tdm_load_tile(Elo + rowA0 * D_DIM, As_lo);
    tdm_load_tile(Ehi + rowB0 * D_DIM, Bs_hi);
    tdm_load_tile(Elo + rowB0 * D_DIM, Bs_lo);
    __builtin_amdgcn_s_wait_tensorcnt(0);
  }
  __syncthreads();

  v8f acc[2][4] = {};  // 8 independent 16x16 f32 accumulators

  for (int kk = 0; kk < D_DIM; kk += 32) {
    v16bf ah[2], al[2], bh[4], bl[4];
#pragma unroll
    for (int i = 0; i < 2; ++i) {
      ah[i] = load_frag(As_hi, wm * 32 + 16 * i, kk, lane);
      al[i] = load_frag(As_lo, wm * 32 + 16 * i, kk, lane);
    }
#pragma unroll
    for (int j = 0; j < 4; ++j) {
      bh[j] = load_frag(Bs_hi, wn * 64 + 16 * j, kk, lane);
      bl[j] = load_frag(Bs_lo, wn * 64 + 16 * j, kk, lane);
    }
    // hi*hi
#pragma unroll
    for (int i = 0; i < 2; ++i)
#pragma unroll
      for (int j = 0; j < 4; ++j)
        acc[i][j] = __builtin_amdgcn_wmma_f32_16x16x32_bf16(
            false, ah[i], false, bh[j], (short)0, acc[i][j], false, false);
    // hi*lo
#pragma unroll
    for (int i = 0; i < 2; ++i)
#pragma unroll
      for (int j = 0; j < 4; ++j)
        acc[i][j] = __builtin_amdgcn_wmma_f32_16x16x32_bf16(
            false, ah[i], false, bl[j], (short)0, acc[i][j], false, false);
    // lo*hi
#pragma unroll
    for (int i = 0; i < 2; ++i)
#pragma unroll
      for (int j = 0; j < 4; ++j)
        acc[i][j] = __builtin_amdgcn_wmma_f32_16x16x32_bf16(
            false, al[i], false, bh[j], (short)0, acc[i][j], false, false);
  }

  // ---- thread-local logsumexp over this thread's 64 tile entries ----
  float m = -3.4e38f;
#pragma unroll
  for (int i = 0; i < 2; ++i)
#pragma unroll
    for (int j = 0; j < 4; ++j)
#pragma unroll
      for (int e = 0; e < 8; ++e) m = fmaxf(m, KAPPA * acc[i][j][e]);
  float s = 0.0f;
#pragma unroll
  for (int i = 0; i < 2; ++i)
#pragma unroll
    for (int j = 0; j < 4; ++j)
#pragma unroll
      for (int e = 0; e < 8; ++e) s += __expf(KAPPA * acc[i][j][e] - m);

  // ---- block reduce (m, s) ----
  red_m[tid] = m;
  red_s[tid] = s;
  __syncthreads();
  for (int off = 128; off > 0; off >>= 1) {
    if (tid < off) {
      float m1 = red_m[tid], s1 = red_s[tid];
      float m2 = red_m[tid + off], s2 = red_s[tid + off];
      float mm = fmaxf(m1, m2);
      red_m[tid] = mm;
      red_s[tid] = s1 * __expf(m1 - mm) + s2 * __expf(m2 - mm);
    }
    __syncthreads();
  }
  if (tid == 0)
    partials[blockIdx.y * gridDim.x + blockIdx.x] =
        make_float2(red_m[0], red_s[0]);
}

// ---------------------------------------------------------------------------
// Kernel 3: combine all block partials -> scalar loss
// ---------------------------------------------------------------------------
__global__ __launch_bounds__(256) void lse_finalize_kernel(
    const float2* __restrict__ partials, int n, float* __restrict__ out) {
  __shared__ float red_m[256];
  __shared__ float red_s[256];
  float m = -3.4e38f, s = 0.0f;
  for (int i = threadIdx.x; i < n; i += 256) {
    float2 p = partials[i];
    float mm = fmaxf(m, p.x);
    s = s * __expf(m - mm) + p.y * __expf(p.x - mm);
    m = mm;
  }
  int tid = threadIdx.x;
  red_m[tid] = m;
  red_s[tid] = s;
  __syncthreads();
  for (int off = 128; off > 0; off >>= 1) {
    if (tid < off) {
      float m1 = red_m[tid], s1 = red_s[tid];
      float m2 = red_m[tid + off], s2 = red_s[tid + off];
      float mm = fmaxf(m1, m2);
      red_m[tid] = mm;
      red_s[tid] = s1 * __expf(m1 - mm) + s2 * __expf(m2 - mm);
    }
    __syncthreads();
  }
  if (tid == 0) out[0] = red_m[0] + logf(red_s[0]);
}

// ---------------------------------------------------------------------------
// Host entry
// ---------------------------------------------------------------------------
extern "C" void kernel_launch(void* const* d_in, const int* in_sizes, int n_in,
                              void* d_out, int out_size, void* d_ws,
                              size_t ws_size, hipStream_t stream) {
  (void)in_sizes; (void)n_in; (void)out_size; (void)ws_size;
  const float* E = (const float*)d_in[0];
  float* out = (float*)d_out;

  const size_t nelem = (size_t)N_PTS * D_DIM;
  unsigned short* hi = (unsigned short*)d_ws;
  unsigned short* lo = hi + nelem;
  float2* partials = (float2*)((char*)d_ws + 2 * nelem * sizeof(unsigned short));

  const int n4 = (int)(nelem / 4);
  split_bf16_kernel<<<(n4 + 255) / 256, 256, 0, stream>>>(E, hi, lo, n4);

  dim3 grid(N_PTS / BN, N_PTS / BM);
  gram_lse_kernel<<<grid, 256, 0, stream>>>(hi, lo, partials);

  lse_finalize_kernel<<<1, 256, 0, stream>>>(
      partials, (N_PTS / BM) * (N_PTS / BN), out);
}
